// SASRec_69183333204242
// MI455X (gfx1250) — matrix operand order repaired and checked by reading, not compiled, not run on hardware
//
#include <hip/hip_runtime.h>
#include <hip/hip_bf16.h>
#include <math.h>

// ---------------------------------------------------------------------------
// SASRec forward + loss for MI455X (gfx1250)
// wave32, v_wmma_f32_16x16x32_bf16, async global->LDS staging for attention
// ---------------------------------------------------------------------------

namespace {

constexpr int   kB     = 256;
constexpr int   kL     = 200;
constexpr int   kD     = 128;
constexpr int   kH     = 4;
constexpr int   kHD    = 32;          // head dim
constexpr int   kNB    = 2;
constexpr int   kItems = 50001;       // ITEMS + 1
constexpr float kEps   = 1e-8f;
constexpr float kC     = 0.001f;
constexpr float kReg   = 0.0001f;
constexpr int   kBL    = kB * kL;     // 51200 tokens
constexpr int   kLP    = 13 * 16;     // 208: L padded to tile multiple
constexpr int   kCP    = 224;         // padded to 32-multiple for PV K-loop

typedef __bf16 bf16_t;
typedef bf16_t v16bf __attribute__((ext_vector_type(16)));
typedef float  v8f   __attribute__((ext_vector_type(8)));
typedef int    v4i_t __attribute__((ext_vector_type(4)));

// ---- WMMA fragment helpers (wave32; contiguous k = 16*(lane/16) + i, same for A and B) ----------

// A-matrix 16x32 bf16 tile from row-major f32 source (lda = row stride).
__device__ __forceinline__ v16bf load_a_f32(const float* A, int lda,
                                            int m0, int k0, int lane) {
  int m  = m0 + (lane & 15);
  int kb = k0 + ((lane & 16) ? 16 : 0);
  const float* p = A + (size_t)m * lda + kb;
  v16bf a;
#pragma unroll
  for (int i = 0; i < 16; ++i) a[i] = (bf16_t)p[i];
  return a;
}

// B-matrix 32x16 bf16 tile for C = A @ W^T with W row-major (N x K):
// B[k][n] = W[n*ldw + k]; per-lane: fixed n, contiguous K run of 16.
__device__ __forceinline__ v16bf load_b_wT(const float* W, int ldw,
                                           int n0, int k0, int lane) {
  int n  = n0 + (lane & 15);
  int kb = k0 + ((lane & 16) ? 16 : 0);
  const float* p = W + (size_t)n * ldw + kb;
  v16bf b;
#pragma unroll
  for (int i = 0; i < 16; ++i) b[i] = (bf16_t)p[i];
  return b;
}

// B-matrix 32x16 bf16 tile where B[k][n] = V[k*ldv + n]  (K-major source).
__device__ __forceinline__ v16bf load_b_kmaj(const float* V, int ldv,
                                             int n0, int k0, int lane) {
  int n  = n0 + (lane & 15);
  int kb = k0 + ((lane & 16) ? 16 : 0);
  v16bf b;
#pragma unroll
  for (int i = 0; i < 16; ++i) b[i] = (bf16_t)V[(size_t)(kb + i) * ldv + n];
  return b;
}

__device__ __forceinline__ v8f wmma_bf16(v16bf a, v16bf b, v8f c) {
  // 8 args: (neg_a, A, neg_b, B, c_mod, C, reuse_a, reuse_b)
  return __builtin_amdgcn_wmma_f32_16x16x32_bf16(false, a, false, b,
                                                 (short)0, c, false, false);
}

// ---- async global -> LDS staging (gfx1250 ASYNCcnt path) -------------------

#if defined(__gfx1250__) && __has_builtin(__builtin_amdgcn_global_load_async_to_lds_b128)
#define USE_ASYNC_LDS 1
#else
#define USE_ASYNC_LDS 0
#endif

// Copy rows x 32 f32 (row stride src_ld floats) into LDS tile (row stride 32).
__device__ __forceinline__ void stage_tile(const float* gsrc, int src_ld,
                                           float* ldst, int rows) {
  int nvec = rows * 8;                      // 16-byte chunks
  for (int q = (int)threadIdx.x; q < nvec; q += (int)blockDim.x) {
    int r = q >> 3, c = (q & 7) * 4;
    const float* g = gsrc + (size_t)r * src_ld + c;
    float* l = ldst + r * kHD + c;
#if USE_ASYNC_LDS
    // Param types per clang diag: (v4i addrspace(1)*, v4i addrspace(3)*, imm, imm)
    __builtin_amdgcn_global_load_async_to_lds_b128(
        (__attribute__((address_space(1))) v4i_t*)g,
        (__attribute__((address_space(3))) v4i_t*)l, 0, 0);
#else
    *(float4*)l = *(const float4*)g;
#endif
  }
}

__device__ __forceinline__ void wait_async_all() {
#if defined(__gfx1250__) && __has_builtin(__builtin_amdgcn_s_wait_asynccnt)
  __builtin_amdgcn_s_wait_asynccnt(0);
#else
  asm volatile("s_wait_asynccnt 0" ::: "memory");
#endif
}

// ---- reductions: wave32 butterfly + 4-wave LDS combine ---------------------

__device__ __forceinline__ float wave_reduce(float v) {
#pragma unroll
  for (int m = 16; m > 0; m >>= 1) v += __shfl_xor(v, m, 32);
  return v;
}

__device__ __forceinline__ float block_reduce128(float v, float* red) {
  int wave = (int)threadIdx.x >> 5, lane = (int)threadIdx.x & 31;
  v = wave_reduce(v);
  if (lane == 0) red[wave] = v;
  __syncthreads();
  float r = (red[0] + red[1]) + (red[2] + red[3]);
  __syncthreads();
  return r;
}

// ---- kernels ---------------------------------------------------------------

// seqs = item_emb[log] + pos_emb[(l+1)*(log!=0)]
__global__ void embed_kernel(const int* __restrict__ logs,
                             const float* __restrict__ item_emb,
                             const float* __restrict__ pos_emb,
                             float* __restrict__ F) {
  int t = blockIdx.x;          // token index
  int d = threadIdx.x;         // feature
  int l = t % kL;
  int it = logs[t];
  int ps = (it != 0) ? (l + 1) : 0;
  F[(size_t)t * kD + d] = item_emb[(size_t)it * kD + d]
                        + pos_emb[(size_t)ps * kD + d];
}

// row LayerNorm over D=128 (one block per token)
__global__ void ln_kernel(const float* __restrict__ X,
                          const float* __restrict__ g,
                          const float* __restrict__ bta,
                          float* __restrict__ Y) {
  __shared__ float red[4];
  int t = blockIdx.x, tid = threadIdx.x;
  float x = X[(size_t)t * kD + tid];
  float m = block_reduce128(x, red) * (1.0f / kD);
  float d = x - m;
  float v = block_reduce128(d * d, red) * (1.0f / kD);
  Y[(size_t)t * kD + tid] = d * rsqrtf(v + kEps) * g[tid] + bta[tid];
}

// C = [relu]( A @ W^T + bias ) [+ R]; A: MxK f32, W: NxK f32.
// Each wave32 computes a 16x64 strip: one A fragment feeds 4 WMMAs.
template <int RELU, int RESID>
__global__ void gemm_kernel(const float* __restrict__ A,
                            const float* __restrict__ W,
                            const float* __restrict__ bias,
                            const float* __restrict__ R,
                            float* __restrict__ Cm,
                            int M, int N, int K) {
  int wave = threadIdx.x >> 5;
  int lane = threadIdx.x & 31;
  int tiles_n = N >> 6;                 // 64-wide strips
  int tiles   = (M >> 4) * tiles_n;
  int tile = blockIdx.x * 8 + wave;
  if (tile >= tiles) return;            // wave-uniform: EXEC stays all-ones
  int tm = tile / tiles_n;
  int tn = tile - tm * tiles_n;
  v8f acc[4] = {};
  for (int k0 = 0; k0 < K; k0 += 32) {
    v16bf a = load_a_f32(A, K, tm * 16, k0, lane);
#pragma unroll
    for (int j = 0; j < 4; ++j) {
      v16bf b = load_b_wT(W, K, tn * 64 + j * 16, k0, lane);
      acc[j] = wmma_bf16(a, b, acc[j]);
    }
  }
  int mb = tm * 16 + ((lane & 16) ? 8 : 0);
#pragma unroll
  for (int j = 0; j < 4; ++j) {
    int n = tn * 64 + j * 16 + (lane & 15);
    float bn = bias[n];
#pragma unroll
    for (int r = 0; r < 8; ++r) {
      float v = acc[j][r] + bn;
      if (RELU)  v = fmaxf(v, 0.0f);
      size_t off = (size_t)(mb + r) * N + n;
      if (RESID) v += R[off];
      Cm[off] = v;
    }
  }
}

// Fused causal attention for one (batch, head).
// LDS: 208x224 f32 scores + q/k/v tiles staged once via async global->LDS
// (262 KB total < 320 KB WGP LDS). attn may alias qb: q is fully staged into
// LDS behind a barrier before any global store happens.
__global__ void attention_kernel(const float* qb, const float* kb,
                                 const float* vb, float* attn) {
  extern __shared__ float smem[];
  float* sc = smem;                          // [208][224] scores/probs
  float* qt = sc + kLP * kCP;                // [208][32]
  float* kt = qt + kLP * kHD;                // [208][32]
  float* vt = kt + kLP * kHD;                // [224][32]

  int bh = blockIdx.x;
  int b = bh >> 2, h = bh & 3;
  const float* qh = qb + (size_t)b * kL * kD + h * kHD;
  const float* kh = kb + (size_t)b * kL * kD + h * kHD;
  const float* vh = vb + (size_t)b * kL * kD + h * kHD;
  int wave = threadIdx.x >> 5, lane = threadIdx.x & 31;
  const float scale = 0.17677669529663687f;  // 1/sqrt(32)

  // Stage q/k/v head slices into LDS (ASYNCcnt-tracked on gfx1250).
  stage_tile(qh, kD, qt, kLP);
  stage_tile(kh, kD, kt, kLP);
  stage_tile(vh, kD, vt, kCP);               // extra rows pad the PV K-loop
  wait_async_all();
  __syncthreads();

  // Phase A: scores = scale * q @ k^T, lower triangle of 13x13 tiles only
  int idx = 0;
  for (int ti = 0; ti < 13; ++ti)
    for (int tj = 0; tj <= ti; ++tj, ++idx)
      if ((idx & 7) == wave) {
        v16bf a  = load_a_f32(qt, kHD, ti * 16, 0, lane);   // K = hd = 32
        v16bf bb = load_b_wT(kt, kHD, tj * 16, 0, lane);
        v8f acc = {};
        acc = wmma_bf16(a, bb, acc);
        int n  = tj * 16 + (lane & 15);
        int mb = ti * 16 + ((lane & 16) ? 8 : 0);
#pragma unroll
        for (int r = 0; r < 8; ++r) sc[(mb + r) * kCP + n] = acc[r] * scale;
      }
  __syncthreads();

  // Phase B: causal softmax per valid row; zero masked/padded columns
  for (int s = (int)threadIdx.x; s < kL; s += (int)blockDim.x) {
    float* row = sc + s * kCP;
    float mx = -3.0e38f;
    for (int t = 0; t <= s; ++t) mx = fmaxf(mx, row[t]);
    float sum = 0.0f;
    for (int t = 0; t <= s; ++t) { float e = __expf(row[t] - mx); row[t] = e; sum += e; }
    float inv = 1.0f / sum;
    for (int t = 0; t <= s; ++t) row[t] *= inv;
    for (int t = s + 1; t < kCP; ++t) row[t] = 0.0f;
  }
  __syncthreads();

  // Phase C: out = P @ V, entirely LDS-fed (K padded to 224, pads zeroed)
  int idx2 = 0;
  for (int ti = 0; ti < 13; ++ti)
    for (int tj = 0; tj < 2; ++tj, ++idx2)
      if ((idx2 & 7) == wave) {
        v8f acc = {};
        for (int k0 = 0; k0 < kCP; k0 += 32) {
          v16bf a  = load_a_f32(sc, kCP, ti * 16, k0, lane);
          v16bf bb = load_b_kmaj(vt, kHD, tj * 16, k0, lane);
          acc = wmma_bf16(a, bb, acc);
        }
        int n  = tj * 16 + (lane & 15);
        int mb = ti * 16 + ((lane & 16) ? 8 : 0);
#pragma unroll
        for (int r = 0; r < 8; ++r) {
          int m = mb + r;
          if (m < kL)
            attn[((size_t)b * kL + m) * kD + h * kHD + n] = acc[r];
        }
      }
}

// u[e] += sum_k E[k,e] * (E[k,:]·w)   (== (E^T E) @ w, streamed)
// scal[0] += sum(E^2)                  (Frobenius reg term)
__global__ void uE_kernel(const float* __restrict__ E,
                          const float* __restrict__ w,
                          float* __restrict__ u,
                          float* __restrict__ scal) {
  __shared__ float red[4];
  int tid = threadIdx.x;
  float wd = w[tid];
  float pu = 0.0f, pss = 0.0f;
  for (int k = blockIdx.x; k < kItems; k += gridDim.x) {
    float e = E[(size_t)k * kD + tid];
    float t = block_reduce128(e * wd, red);   // E[k]·w
    pu  += t * e;
    pss += e * e;
  }
  atomicAdd(&u[tid], pu);
  float ss = block_reduce128(pss, red);
  if (tid == 0) atomicAdd(&scal[0], ss);
}

// Rank-1 collapse of hh = w^T w:
//   left_b  = S * C * sum_s rowsum(F[b,s]) * (F[b,s]·u),  u = EE@w, S = sum w
//   right_b = sum_s ((1-C) R^2 - 2 R),  R = sum_d w_d F[b,s,d] E[pos[b,s],d]
__global__ void loss_kernel(const float* __restrict__ F,
                            const float* __restrict__ E,
                            const int* __restrict__ pos,
                            const float* __restrict__ w,
                            const float* __restrict__ u,
                            float* __restrict__ scal) {
  __shared__ float red[4];
  int b = blockIdx.x, tid = threadIdx.x;
  float wd = w[tid];
  float S = block_reduce128(wd, red);
  float ud = u[tid];
  float left = 0.0f, right = 0.0f;
  for (int s = 0; s < kL; ++s) {
    float f  = F[((size_t)b * kL + s) * kD + tid];
    int   pi = pos[b * kL + s];
    float pe = E[(size_t)pi * kD + tid];
    float rs = block_reduce128(f, red);
    float du = block_reduce128(f * ud, red);
    float R  = block_reduce128(wd * f * pe, red);
    left  += rs * du;
    right += (1.0f - kC) * R * R - 2.0f * R;
  }
  if (tid == 0) atomicAdd(&scal[1], S * kC * left + right);
}

__global__ void finalize_kernel(const float* __restrict__ scal,
                                float* __restrict__ out) {
  out[0] = scal[1] * (1.0f / kB) + kReg * sqrtf(scal[0]);
}

} // anonymous namespace

// ---------------------------------------------------------------------------

extern "C" void kernel_launch(void* const* d_in, const int* in_sizes, int n_in,
                              void* d_out, int out_size, void* d_ws, size_t ws_size,
                              hipStream_t stream) {
  (void)in_sizes; (void)n_in; (void)out_size; (void)ws_size;

  const int*   log_seqs = (const int*)  d_in[1];
  const int*   pos_seqs = (const int*)  d_in[2];
  const float* item_emb = (const float*)d_in[3];
  const float* pos_emb  = (const float*)d_in[4];
  const float* pred_w   = (const float*)d_in[5];
  const float* ln1_g    = (const float*)d_in[6];
  const float* ln1_b    = (const float*)d_in[7];
  const float* in_w     = (const float*)d_in[8];
  const float* in_b     = (const float*)d_in[9];
  const float* out_w    = (const float*)d_in[10];
  const float* out_b    = (const float*)d_in[11];
  const float* ln2_g    = (const float*)d_in[12];
  const float* ln2_b    = (const float*)d_in[13];
  const float* c1_w     = (const float*)d_in[14];
  const float* c1_b     = (const float*)d_in[15];
  const float* c2_w     = (const float*)d_in[16];
  const float* c2_b     = (const float*)d_in[17];

  // Workspace: 5 padded (B,L,D) f32 buffers (~131 MB) + tiny accumulators.
  // Pads absorb benign over-reads from attention's 16-tile rounding.
  const size_t BUFE = (size_t)kBL * kD + 8192;
  float* ws   = (float*)d_ws;
  float* F    = ws;            // running sequence state
  float* Qln  = F    + BUFE;   // LayerNorm output / residual source
  float* qb   = Qln  + BUFE;   // q projections; attention output aliases this
  float* kbuf = qb   + BUFE;   // k projections; FFN hidden reuses this
  float* vbuf = kbuf + BUFE;   // v projections
  float* u    = vbuf + BUFE;   // (EE @ w) accumulator, 128 floats
  float* scal = u + kD;        // [0]=sum(E^2), [1]=loss accumulator

  (void)hipMemsetAsync(u, 0, (kD + 8) * sizeof(float), stream);

  embed_kernel<<<dim3(kBL), dim3(kD), 0, stream>>>(log_seqs, item_emb, pos_emb, F);

  const int gemm_grid = ((kBL / 16) * (kD / 64) + 7) / 8;
  const size_t attn_lds =
      (size_t)(kLP * kCP + 2 * kLP * kHD + kCP * kHD) * sizeof(float); // 268 288 B

  for (int i = 0; i < kNB; ++i) {
    const float* Wqkv = in_w + (size_t)i * 3 * kD * kD;
    const float* bqkv = in_b + (size_t)i * 3 * kD;

    ln_kernel<<<dim3(kBL), dim3(kD), 0, stream>>>(F, ln1_g + i * kD, ln1_b + i * kD, Qln);

    gemm_kernel<0,0><<<gemm_grid, 256, 0, stream>>>(Qln, Wqkv,               bqkv,          nullptr, qb,   kBL, kD, kD);
    gemm_kernel<0,0><<<gemm_grid, 256, 0, stream>>>(F,   Wqkv + kD * kD,     bqkv + kD,     nullptr, kbuf, kBL, kD, kD);
    gemm_kernel<0,0><<<gemm_grid, 256, 0, stream>>>(F,   Wqkv + 2 * kD * kD, bqkv + 2 * kD, nullptr, vbuf, kBL, kD, kD);

    attention_kernel<<<dim3(kB * kH), 256, attn_lds, stream>>>(qb, kbuf, vbuf, qb);

    // seqs = Qln + attn @ out_w^T + out_b   (residual on Q, per source)
    gemm_kernel<0,1><<<gemm_grid, 256, 0, stream>>>(qb, out_w + (size_t)i * kD * kD, out_b + i * kD, Qln, F, kBL, kD, kD);

    ln_kernel<<<dim3(kBL), dim3(kD), 0, stream>>>(F, ln2_g + i * kD, ln2_b + i * kD, Qln);

    // FFN: seqs = ln2 + relu(ln2 @ c1^T + c1_b) @ c2^T + c2_b
    gemm_kernel<1,0><<<gemm_grid, 256, 0, stream>>>(Qln,  c1_w + (size_t)i * kD * kD, c1_b + i * kD, nullptr, kbuf, kBL, kD, kD);
    gemm_kernel<0,1><<<gemm_grid, 256, 0, stream>>>(kbuf, c2_w + (size_t)i * kD * kD, c2_b + i * kD, Qln,     F,    kBL, kD, kD);
  }

  uE_kernel  <<<dim3(256), dim3(kD), 0, stream>>>(item_emb, pred_w, u, scal);
  loss_kernel<<<dim3(kB),  dim3(kD), 0, stream>>>(F, item_emb, pos_seqs, pred_w, u, scal);
  finalize_kernel<<<1, 1, 0, stream>>>(scal, (float*)d_out);
}
